// Qwen3OmniMoeAttention_88424786690397
// MI455X (gfx1250) — matrix operand order repaired and checked by reading, not compiled, not run on hardware
//
#include <hip/hip_runtime.h>

// ---------------------------------------------------------------------------
// Qwen3 GQA attention layer for MI455X (gfx1250, wave32, WMMA).
// - All matmuls (QKV proj, QK^T, PV, out proj) on v_wmma_f32_16x16x32_bf16
//   with fp32 accumulation (fp32 WMMA is only K=4; bf16 is the fast path).
// - Weights pre-transposed once so every WMMA B-fragment is a contiguous load.
// - Attention: 8 waves/block share one head's K/V -> cooperative double-
//   buffered GLOBAL_LOAD_ASYNC_TO_LDS_B128 staging (ASYNCcnt), fragments
//   served from LDS. 192MB L2 keeps all bf16 operands resident.
// ---------------------------------------------------------------------------

typedef __bf16 bf16_t;
typedef __attribute__((ext_vector_type(16))) __bf16 v16bf;
typedef __attribute__((ext_vector_type(8)))  __bf16 v8bf;
typedef __attribute__((ext_vector_type(8)))  float  v8f;

#define S_LEN   4096
#define HID     2048
#define HQ      32
#define HKV     4
#define DHEAD   128
#define QDIM    (HQ * DHEAD)    // 4096
#define KVDIM   (HKV * DHEAD)   // 512

// LDS tile geometry (padded so 16-lane column reads hit distinct banks)
#define KT_LD    136                   // 128 d + 8 pad (row stride, elems)
#define KT_ELEMS (32 * KT_LD)          // 4352
#define VT_LD    40                    // 32 keys + 8 pad
#define VT_ELEMS (128 * VT_LD)         // 5120
#define PST_LD   40

// ---- bf16 helpers (round-to-nearest-even) ---------------------------------
__device__ inline bf16_t f2bf(float f) {
  unsigned u = __builtin_bit_cast(unsigned, f);
  u += 0x7FFFu + ((u >> 16) & 1u);
  unsigned short h = (unsigned short)(u >> 16);
  return __builtin_bit_cast(bf16_t, h);
}
__device__ inline float bf2f(bf16_t b) {
  unsigned short s = __builtin_bit_cast(unsigned short, b);
  unsigned u = ((unsigned)s) << 16;
  return __builtin_bit_cast(float, u);
}
__device__ inline void store_val(bf16_t* p, float v) { *p = f2bf(v); }
__device__ inline void store_val(float*  p, float v) { *p = v; }

// ---- WMMA fragment loaders (layouts per CDNA5 ISA 7.12.2) -----------------
// A 16x32 bf16, row-major src, leading dim lda.
__device__ inline v16bf load_a16x32(const bf16_t* __restrict__ src, int lda) {
  const int lane = threadIdx.x & 31;
  const int row  = lane & 15;
  const int kb   = (lane >> 4) << 3;           // 0 or 8
  const bf16_t* p = src + (long)row * lda + kb;
  v8bf lo = *(const v8bf*)p;
  v8bf hi = *(const v8bf*)(p + 16);
  v16bf a;
#pragma unroll
  for (int i = 0; i < 8; ++i) { a[i] = lo[i]; a[8 + i] = hi[i]; }
  return a;
}

// B 32x16 where B = X^T, X row-major [n][k], leading dim ldx. Two 16B loads/lane (k=kb..kb+7, kb+16..kb+23).
__device__ inline v16bf load_bT(const bf16_t* __restrict__ X, int ldx) {
  const int lane = threadIdx.x & 31;
  const int n    = lane & 15;
  const int kb   = (lane >> 4) << 3;           // 0 or 8
  const bf16_t* p = X + (long)n * ldx + kb;
  v8bf lo = *(const v8bf*)p;
  v8bf hi = *(const v8bf*)(p + 16);
  v16bf b;
#pragma unroll
  for (int i = 0; i < 8; ++i) { b[i] = lo[i]; b[8 + i] = hi[i]; }
  return b;
}

__device__ inline v8f wmma_bf16(v16bf a, v16bf b, v8f c) {
  return __builtin_amdgcn_wmma_f32_16x16x32_bf16(
      false, a, false, b, (short)0, c, false, false);
}

// ---- async global -> LDS copy (CDNA5, tracked by ASYNCcnt) ----------------
__device__ inline void async_ld_b128(unsigned lds_off, const bf16_t* gaddr) {
  asm volatile("global_load_async_to_lds_b128 %0, %1, off"
               :: "v"(lds_off), "v"(gaddr) : "memory");
}
__device__ inline void wait_asynccnt0() {
  asm volatile("s_wait_asynccnt 0" ::: "memory");
}

// ---- fp32 -> bf16 conversion ----------------------------------------------
__global__ void cvt_kernel(const float* __restrict__ in, bf16_t* __restrict__ out, long n) {
  long i = (long)blockIdx.x * blockDim.x + threadIdx.x;
  if (i < n) out[i] = f2bf(in[i]);
}

// in [R][C] row-major fp32 -> out [C][R] bf16 (transpose)
__global__ void cvtT_kernel(const float* __restrict__ in, bf16_t* __restrict__ out,
                            int R, int C) {
  long i = (long)blockIdx.x * blockDim.x + threadIdx.x;
  if (i >= (long)R * C) return;
  int r = (int)(i / C), c = (int)(i % C);
  out[(long)c * R + r] = f2bf(in[i]);
}

// ---- RoPE (rotate-half, wavelength 10000) ---------------------------------
__global__ void rope_kernel(const bf16_t* __restrict__ in, bf16_t* __restrict__ out,
                            const int* __restrict__ pos, int nh) {
  long idx = (long)blockIdx.x * blockDim.x + threadIdx.x;
  long total = (long)S_LEN * nh * 64;
  if (idx >= total) return;
  int d = (int)(idx & 63);
  int h = (int)((idx >> 6) % nh);
  int s = (int)(idx / (64L * nh));
  float p = (float)pos[s];
  float inv = __expf(-((float)d * (1.0f / 64.0f)) * 9.210340371976184f);
  float f = p * inv, sn, cs;
  __sincosf(f, &sn, &cs);
  long base = ((long)s * nh + h) * DHEAD + d;
  float x1 = bf2f(in[base]);
  float x2 = bf2f(in[base + 64]);
  out[base]      = f2bf(x1 * cs - x2 * sn);
  out[base + 64] = f2bf(x2 * cs + x1 * sn);
}

// ---- Tiled bf16 GEMM: C[M][N] = A[M][K] * B[K][N], B given transposed -----
template <typename CT, bool TRANS>
__global__ __launch_bounds__(256) void gemm_kernel(
    const bf16_t* __restrict__ A, const bf16_t* __restrict__ BT,
    CT* __restrict__ C, int M, int N, int K) {
  const int lane  = threadIdx.x & 31;
  const int wave  = threadIdx.x >> 5;
  const int bm    = blockIdx.x * 128 + (wave >> 2) * 64;
  const int bn    = blockIdx.y * 128 + (wave & 3) * 32;

  v8f acc[4][2];
#pragma unroll
  for (int i = 0; i < 4; ++i)
#pragma unroll
    for (int j = 0; j < 2; ++j)
#pragma unroll
      for (int e = 0; e < 8; ++e) acc[i][j][e] = 0.f;

  for (int k0 = 0; k0 < K; k0 += 32) {
    if (k0 + 32 < K) {
      __builtin_prefetch(A  + (long)bm * K + k0 + 32, 0, 0);
      __builtin_prefetch(BT + (long)bn * K + k0 + 32, 0, 0);
    }
    v16bf a[4], b[2];
#pragma unroll
    for (int i = 0; i < 4; ++i)
      a[i] = load_a16x32(A + (long)(bm + i * 16) * K + k0, K);
#pragma unroll
    for (int j = 0; j < 2; ++j)
      b[j] = load_bT(BT + (long)(bn + j * 16) * K + k0, K);
#pragma unroll
    for (int i = 0; i < 4; ++i)
#pragma unroll
      for (int j = 0; j < 2; ++j)
        acc[i][j] = wmma_bf16(a[i], b[j], acc[i][j]);
  }

  const int col = lane & 15;
  const int rb  = (lane >> 4) << 3;
#pragma unroll
  for (int i = 0; i < 4; ++i)
#pragma unroll
    for (int j = 0; j < 2; ++j) {
      int r0 = bm + i * 16 + rb;
      int c0 = bn + j * 16 + col;
#pragma unroll
      for (int e = 0; e < 8; ++e) {
        if (TRANS) store_val(&C[(long)c0 * M + (r0 + e)], acc[i][j][e]);
        else       store_val(&C[(long)(r0 + e) * N + c0], acc[i][j][e]);
      }
    }
}

// ---- Flash attention (causal, GQA 8:1), async-staged K/V ------------------
// Block = 8 waves, one head, 128 consecutive queries (16 per wave).
// K/V tiles for each 32-key block are DMA'd into LDS (double buffered) by
// all 256 threads via global_load_async_to_lds_b128 and shared by all waves.
__global__ __launch_bounds__(256) void attn_kernel(
    const bf16_t* __restrict__ Q, const bf16_t* __restrict__ Kr,
    const bf16_t* __restrict__ Vt, bf16_t* __restrict__ O) {
  __shared__ __align__(16) bf16_t smem[2 * KT_ELEMS + 2 * VT_ELEMS + 8 * 16 * PST_LD];

  const int tid   = threadIdx.x;
  const int lane  = tid & 31;
  const int wave  = tid >> 5;
  const int h     = blockIdx.y;
  const int kvh   = h >> 3;                 // HQ/HKV = 8
  const int qblk  = blockIdx.x * 128;
  const int qbase = qblk + wave * 16;
  const int kend  = qblk + 128;             // uniform trip count in block
  const int col   = lane & 15;
  const int rb    = (lane >> 4) << 3;
  const float scale = 0.08838834764831843f; // 1/sqrt(128)

  bf16_t* pst = smem + 2 * KT_ELEMS + 2 * VT_ELEMS + wave * 16 * PST_LD;

  // LDS byte offset of smem (runtime addrspacecast+truncate; no static init)
  unsigned smem_base = (unsigned)(size_t)(void*)smem;

  // ---- Q A-fragments (4 K-chunks of d) ----
  const bf16_t* qptr = Q + (long)qbase * QDIM + h * DHEAD;
  v16bf qa[4];
#pragma unroll
  for (int c = 0; c < 4; ++c) qa[c] = load_a16x32(qptr + c * 32, QDIM);

  float m[8], l[8];
  v8f o[8];
#pragma unroll
  for (int i = 0; i < 8; ++i) { m[i] = -1e30f; l[i] = 0.f; }
#pragma unroll
  for (int t = 0; t < 8; ++t)
#pragma unroll
    for (int e = 0; e < 8; ++e) o[t][e] = 0.f;

  // cooperative async fill of one (K,V) 32-key tile pair into buffer `buf`
  auto fill = [&](int buf, int kb) {
    unsigned kbase = smem_base + (unsigned)buf * (KT_ELEMS * 2u);
    unsigned vbase = smem_base + 2u * (KT_ELEMS * 2u) + (unsigned)buf * (VT_ELEMS * 2u);
#pragma unroll
    for (int r = 0; r < 2; ++r) {
      int id = tid + r * 256;
      // K tile: 32 rows x 128 d  (16B chunks: 32x16)
      int krow = id >> 4, kseg = id & 15;
      async_ld_b128(kbase + (unsigned)(krow * (KT_LD * 2) + kseg * 16),
                    Kr + (long)(kb + krow) * KVDIM + kvh * DHEAD + kseg * 8);
      // V tile: 128 d-rows x 32 keys (16B chunks: 128x4), from Vt [d][s]
      int vrow = id >> 2, vseg = id & 3;
      async_ld_b128(vbase + (unsigned)(vrow * (VT_LD * 2) + vseg * 16),
                    Vt + (long)(kvh * DHEAD + vrow) * S_LEN + kb + vseg * 8);
    }
  };

  // prologue: stage first tile
  fill(0, 0);
  wait_asynccnt0();
  __syncthreads();

  for (int kb = 0; kb < kend; kb += 32) {
    const int buf = (kb >> 5) & 1;
    if (kb + 32 < kend) fill(buf ^ 1, kb + 32);   // prefetch next tile

    const bf16_t* ktp = smem + buf * KT_ELEMS;
    const bf16_t* vtp = smem + 2 * KT_ELEMS + buf * VT_ELEMS;

    // ---- scores S[16q][32k] as two C fragments ----
    v8f s0, s1;
#pragma unroll
    for (int e = 0; e < 8; ++e) { s0[e] = 0.f; s1[e] = 0.f; }
#pragma unroll
    for (int c = 0; c < 4; ++c) {
      v16bf b0 = load_bT(ktp + c * 32, KT_LD);
      v16bf b1 = load_bT(ktp + 16 * KT_LD + c * 32, KT_LD);
      s0 = wmma_bf16(qa[c], b0, s0);
      s1 = wmma_bf16(qa[c], b1, s1);
    }

    // ---- causal mask + scale (per-lane selects, EXEC stays full) ----
    float p0[8], p1[8], bmax[8];
#pragma unroll
    for (int i = 0; i < 8; ++i) {
      int qrow = qbase + rb + i;
      float v0 = s0[i] * scale;
      float v1 = s1[i] * scale;
      v0 = (kb + col      > qrow) ? -1e30f : v0;
      v1 = (kb + 16 + col > qrow) ? -1e30f : v1;
      p0[i] = v0; p1[i] = v1;
      bmax[i] = fmaxf(v0, v1);
    }
#pragma unroll
    for (int off = 1; off < 16; off <<= 1)
#pragma unroll
      for (int i = 0; i < 8; ++i)
        bmax[i] = fmaxf(bmax[i], __shfl_xor(bmax[i], off, 32));

    float alpha[8], rs[8];
#pragma unroll
    for (int i = 0; i < 8; ++i) {
      float mn = fmaxf(m[i], bmax[i]);
      alpha[i] = __expf(m[i] - mn);
      m[i] = mn;
      p0[i] = __expf(p0[i] - mn);
      p1[i] = __expf(p1[i] - mn);
      rs[i] = p0[i] + p1[i];
    }
#pragma unroll
    for (int off = 1; off < 16; off <<= 1)
#pragma unroll
      for (int i = 0; i < 8; ++i)
        rs[i] += __shfl_xor(rs[i], off, 32);
#pragma unroll
    for (int i = 0; i < 8; ++i) l[i] = l[i] * alpha[i] + rs[i];
#pragma unroll
    for (int t = 0; t < 8; ++t)
#pragma unroll
      for (int i = 0; i < 8; ++i) o[t][i] *= alpha[i];

    // ---- remap P (C layout) -> A layout via per-wave LDS staging ----
#pragma unroll
    for (int i = 0; i < 8; ++i) {
      pst[(rb + i) * PST_LD + col]      = f2bf(p0[i]);
      pst[(rb + i) * PST_LD + 16 + col] = f2bf(p1[i]);
    }
    asm volatile("s_wait_dscnt 0" ::: "memory");   // per-wave DS ordering
    v16bf pa = load_a16x32(pst, PST_LD);

    // ---- O += P * V  (8 d-tiles) ----
#pragma unroll
    for (int t = 0; t < 8; ++t) {
      v16bf vb = load_bT(vtp + t * 16 * VT_LD, VT_LD);
      o[t] = wmma_bf16(pa, vb, o[t]);
    }

    wait_asynccnt0();    // our share of next tile landed
    __syncthreads();     // everyone done filling next / reading current
  }

  // ---- normalize and store ----
  bf16_t* optr = O + (long)qbase * QDIM + h * DHEAD;
#pragma unroll
  for (int t = 0; t < 8; ++t) {
#pragma unroll
    for (int i = 0; i < 8; ++i) {
      float v = o[t][i] / l[i];
      optr[(long)(rb + i) * QDIM + t * 16 + col] = f2bf(v);
    }
  }
}

// ---------------------------------------------------------------------------
extern "C" void kernel_launch(void* const* d_in, const int* in_sizes, int n_in,
                              void* d_out, int out_size, void* d_ws, size_t ws_size,
                              hipStream_t stream) {
  const float* hs = (const float*)d_in[0];
  const float* Wq = (const float*)d_in[1];
  const float* Wk = (const float*)d_in[2];
  const float* Wv = (const float*)d_in[3];
  const float* Wo = (const float*)d_in[4];
  const int*  pos = (const int*)d_in[5];
  // d_in[6] = attention_mask (causal) -- handled analytically
  float* out = (float*)d_out;

  char* w = (char*)d_ws;
  auto take = [&](size_t elems) {
    bf16_t* p = (bf16_t*)w;
    w += elems * sizeof(bf16_t);
    return p;
  };
  bf16_t* hsb = take((size_t)S_LEN * HID);   // hidden bf16          16 MB
  bf16_t* wqT = take((size_t)HID * QDIM);    // Wq^T  [4096][2048]   16 MB
  bf16_t* wkT = take((size_t)HID * KVDIM);   // Wk^T  [512][2048]     2 MB
  bf16_t* wvT = take((size_t)HID * KVDIM);   // Wv^T  [512][2048]     2 MB
  bf16_t* woT = take((size_t)QDIM * HID);    // Wo^T  [2048][4096]   16 MB
  bf16_t* qb  = take((size_t)S_LEN * QDIM);  // Q proj               32 MB
  bf16_t* kbf = take((size_t)S_LEN * KVDIM); // K proj                4 MB
  bf16_t* qr  = take((size_t)S_LEN * QDIM);  // Q roped              32 MB
  bf16_t* kr  = take((size_t)S_LEN * KVDIM); // K roped               4 MB
  bf16_t* vt  = take((size_t)KVDIM * S_LEN); // V^T [512][4096]       4 MB
  bf16_t* ob  = take((size_t)S_LEN * QDIM);  // attn out             32 MB

  const int T = 256;
  long n;

  n = (long)S_LEN * HID;
  cvt_kernel<<<(unsigned)((n + T - 1) / T), T, 0, stream>>>(hs, hsb, n);
  n = (long)HID * QDIM;
  cvtT_kernel<<<(unsigned)((n + T - 1) / T), T, 0, stream>>>(Wq, wqT, HID, QDIM);
  n = (long)HID * KVDIM;
  cvtT_kernel<<<(unsigned)((n + T - 1) / T), T, 0, stream>>>(Wk, wkT, HID, KVDIM);
  cvtT_kernel<<<(unsigned)((n + T - 1) / T), T, 0, stream>>>(Wv, wvT, HID, KVDIM);
  n = (long)QDIM * HID;
  cvtT_kernel<<<(unsigned)((n + T - 1) / T), T, 0, stream>>>(Wo, woT, QDIM, HID);

  // projections
  gemm_kernel<bf16_t, false><<<dim3(S_LEN / 128, QDIM / 128),  T, 0, stream>>>(hsb, wqT, qb,  S_LEN, QDIM,  HID);
  gemm_kernel<bf16_t, false><<<dim3(S_LEN / 128, KVDIM / 128), T, 0, stream>>>(hsb, wkT, kbf, S_LEN, KVDIM, HID);
  gemm_kernel<bf16_t, true ><<<dim3(S_LEN / 128, KVDIM / 128), T, 0, stream>>>(hsb, wvT, vt,  S_LEN, KVDIM, HID);

  // RoPE
  n = (long)S_LEN * HQ * 64;
  rope_kernel<<<(unsigned)((n + T - 1) / T), T, 0, stream>>>(qb, qr, pos, HQ);
  n = (long)S_LEN * HKV * 64;
  rope_kernel<<<(unsigned)((n + T - 1) / T), T, 0, stream>>>(kbf, kr, pos, HKV);

  // flash attention: 128 queries + 1 head per block, async-staged K/V
  attn_kernel<<<dim3(S_LEN / 128, HQ), T, 0, stream>>>(qr, kr, vt, ob);

  // output projection (fp32 out)
  gemm_kernel<float, false><<<dim3(S_LEN / 128, HID / 128), T, 0, stream>>>(ob, woT, out, S_LEN, HID, QDIM);
}
